// MTLPINN_70274254897242
// MI455X (gfx1250) — matrix operand order repaired and checked by reading, not compiled, not run on hardware
//
#include <hip/hip_runtime.h>
#include <hip/hip_bf16.h>

// ---------------------------------------------------------------------------
// Fused multi-task PINN evaluator for MI455X (gfx1250, wave32, WMMA).
//
// Taylor-jet forward propagation: 13 derivative components per neuron.
// Jet order: [0]=val [1]=dx [2]=dy [3]=dt [4]=dxx [5]=dxy [6]=dyy
//            [7]=dxt [8]=dyt [9]=dxxx [10]=dxxy [11]=dxyy [12]=dyyy
// Linear layers -> GEMM over M = 13*16 rows; tanh -> closed-form chain rule.
// Jets never touch HBM: LDS double buffer (~246 KB of CDNA5's 320 KB/WGP).
// Weights pre-converted to f16 in d_ws once per launch (f32 fallback kept).
// Both N-tiles of a wave are accumulated simultaneously so each A fragment
// is read from LDS exactly once (1 ds_load_b128 per v_wmma).
// ---------------------------------------------------------------------------

typedef _Float16 v16h __attribute__((ext_vector_type(16)));
typedef _Float16 v8h  __attribute__((ext_vector_type(8)));
typedef float    v8f  __attribute__((ext_vector_type(8)));

#define NPTS  65536
#define HID   256
#define SPW   16                 // samples per workgroup
#define PITCH 264                // halves per LDS row (256 + 8 pad, keeps b128 alignment)
#define BUF_HALVES (13 * 16 * PITCH)   // 54912 halves = 107.25 KB
#define HD_HALVES  (3 * 16 * PITCH)    // 12672 halves
#define SMEM_BYTES ((2 * BUF_HALVES + HD_HALVES) * 2 + 16 * 17 * 4 + 48 * 4)
#define WMAT (HID * HID)               // 65536 elements per 256x256 matrix
#define WS_NEEDED ((size_t)(4 * WMAT * sizeof(_Float16)))

// tanh via v_exp_f32/v_rcp_f32 (TRANS units; co-executes with XDL WMMA).
// Saturates correctly: x>>0 -> 1, x<<0 -> -1.
__device__ __forceinline__ float fast_tanh(float x) {
    const float e = __expf(2.f * x);
    return 1.f - 2.f / (e + 1.f);
}

template <bool F16W>
__device__ __forceinline__ v16h load_bfrag(const float* __restrict__ Wf,
                                           const _Float16* __restrict__ Wh,
                                           int o, int ks, int half16) {
    v16h bfrag;
    if constexpr (F16W) {
        const v8h* wp = (const v8h*)(Wh + o * HID + ks * 32 + half16 * 16);
        const v8h wlo = wp[0];               // global_load_b128
        const v8h whi = wp[1];               // global_load_b128
#pragma unroll
        for (int e = 0; e < 8; ++e) { bfrag[e] = wlo[e]; bfrag[8 + e] = whi[e]; }
    } else {
        const float4* wp = (const float4*)(Wf + o * HID + ks * 32 + half16 * 16);
#pragma unroll
        for (int q4 = 0; q4 < 4; ++q4) {
            float4 wv = wp[q4];
            bfrag[q4 * 4 + 0] = (_Float16)wv.x;
            bfrag[q4 * 4 + 1] = (_Float16)wv.y;
            bfrag[q4 * 4 + 2] = (_Float16)wv.z;
            bfrag[q4 * 4 + 3] = (_Float16)wv.w;
        }
    }
    return bfrag;
}

// One 256x256 linear layer + tanh applied to an NC-component jet.
// M = NC*16 rows (row = comp*16 + sample), N = 256 outputs, K = 256 inputs.
// 8 waves; each wave accumulates BOTH of its N-tiles at once (acc[NC][2]),
// so each LDS A-fragment feeds two v_wmma issues.
// C-layout => all NC jet components of one (sample,neuron) sit in the same
// lane & accumulator slot, so the tanh jet chain rule is lane-local VALU.
template <int NC, bool F16W>
__device__ __forceinline__ void gemm_tanh_jet(const _Float16* __restrict__ inbuf,
                                              _Float16* __restrict__ outbuf,
                                              const float* __restrict__ Wf,     // f32 weights (fallback)
                                              const _Float16* __restrict__ Wh,  // f16 weights (pre-converted)
                                              const float* __restrict__ bias,
                                              int lane, int wave) {
    const int half16 = lane >> 4;   // which 16-lane half of the wave
    const int nlo    = lane & 15;
    const int o0     = wave * 32 + nlo;        // N-tile 2*wave
    const int o1     = wave * 32 + 16 + nlo;   // N-tile 2*wave+1

    v8f acc[NC][2];
#pragma unroll
    for (int c = 0; c < NC; ++c) {
        v8f z = {};
        acc[c][0] = z;
        acc[c][1] = z;
    }

#pragma unroll
    for (int ks = 0; ks < 8; ++ks) {
        // B fragments (32x16 f16) for both owned N-tiles:
        // lane holds column o, K = ks*32 + half16*16 .. +15.
        const v16h bfrag0 = load_bfrag<F16W>(Wf, Wh, o0, ks, half16);
        const v16h bfrag1 = load_bfrag<F16W>(Wf, Wh, o1, ks, half16);
        // Each A fragment (jet component) loaded once, used by two WMMAs.
#pragma unroll
        for (int c = 0; c < NC; ++c) {
            // A fragment (16x32 f16), contiguous k order, same as B:
            // lane row M = lane%16; halves: K = ks*32 + 16*half16 + [0..15].
            const _Float16* ap = inbuf + (c * 16 + nlo) * PITCH + ks * 32 + half16 * 16;
            const v8h lo = *(const v8h*)ap;          // ds_load_b128
            const v8h hi = *(const v8h*)(ap + 8);    // ds_load_b128
            v16h afrag;
#pragma unroll
            for (int e = 0; e < 8; ++e) { afrag[e] = lo[e]; afrag[8 + e] = hi[e]; }
            acc[c][0] = __builtin_amdgcn_wmma_f32_16x16x32_f16(
                false, afrag, false, bfrag0, (short)0, acc[c][0], false, false);
            acc[c][1] = __builtin_amdgcn_wmma_f32_16x16x32_f16(
                false, afrag, false, bfrag1, (short)0, acc[c][1], false, false);
        }
    }

    // Epilogue: bias (value component only) + tanh jet chain rule, store f16.
#pragma unroll
    for (int nt = 0; nt < 2; ++nt) {
        const int o = nt ? o1 : o0;
        const float bv = bias[o];
#pragma unroll
        for (int r = 0; r < 8; ++r) {
            const int s = r + half16 * 8;       // sample index (C-layout M row)
            float u[NC];
#pragma unroll
            for (int c = 0; c < NC; ++c) u[c] = acc[c][nt][r];
            u[0] += bv;
            const float h  = fast_tanh(u[0]);
            const float d1 = 1.f - h * h;            // tanh'
            float vv[NC];
            vv[0] = h;
            if constexpr (NC == 3) {                 // data head: value + dx,dy only
                vv[1] = d1 * u[1];
                vv[2] = d1 * u[2];
            } else {                                 // full 13-jet (Faa di Bruno to 3rd order)
                const float d2 = -2.f * h * d1;             // tanh''
                const float d3 = d1 * (6.f * h * h - 2.f);  // tanh'''
                vv[1]  = d1 * u[1];
                vv[2]  = d1 * u[2];
                vv[3]  = d1 * u[3];
                vv[4]  = d2 * u[1] * u[1] + d1 * u[4];
                vv[5]  = d2 * u[1] * u[2] + d1 * u[5];
                vv[6]  = d2 * u[2] * u[2] + d1 * u[6];
                vv[7]  = d2 * u[1] * u[3] + d1 * u[7];
                vv[8]  = d2 * u[2] * u[3] + d1 * u[8];
                vv[9]  = d3 * u[1] * u[1] * u[1] + 3.f * d2 * u[1] * u[4] + d1 * u[9];
                vv[10] = d3 * u[1] * u[1] * u[2] + d2 * (2.f * u[5] * u[1] + u[4] * u[2]) + d1 * u[10];
                vv[11] = d3 * u[1] * u[2] * u[2] + d2 * (2.f * u[5] * u[2] + u[6] * u[1]) + d1 * u[11];
                vv[12] = d3 * u[2] * u[2] * u[2] + 3.f * d2 * u[2] * u[6] + d1 * u[12];
            }
#pragma unroll
            for (int c = 0; c < NC; ++c)
                outbuf[(c * 16 + s) * PITCH + o] = (_Float16)vv[c];
        }
    }
}

// Once-per-launch f32 -> f16 weight conversion into workspace.
// Layout in ws: [0]=W2 [1]=W3 [2]=Wp1 [3]=Wd1, each 256x256 row-major.
__global__ __launch_bounds__(256) void convert_weights(const float* __restrict__ W2,
                                                       const float* __restrict__ W3,
                                                       const float* __restrict__ Wp1,
                                                       const float* __restrict__ Wd1,
                                                       _Float16* __restrict__ ws) {
    const int i = blockIdx.x * 256 + threadIdx.x;     // 0 .. 4*65536-1
    const int m = i >> 16;
    const int r = i & (WMAT - 1);
    const float* src = (m == 0) ? W2 : (m == 1) ? W3 : (m == 2) ? Wp1 : Wd1;
    ws[i] = (_Float16)src[r];
}

template <bool F16W>
__global__ __launch_bounds__(256) void pinn_jet_fused(
    const float* __restrict__ x, const float* __restrict__ y, const float* __restrict__ t,
    const float* __restrict__ W1, const float* __restrict__ b1,
    const float* __restrict__ W2, const float* __restrict__ b2,
    const float* __restrict__ W3, const float* __restrict__ b3,
    const float* __restrict__ Wp1, const float* __restrict__ bp1,
    const float* __restrict__ Wp2,
    const float* __restrict__ Wd1, const float* __restrict__ bd1,
    const float* __restrict__ Wd2, const float* __restrict__ bd2,
    const float* __restrict__ lam1p, const float* __restrict__ lam2p,
    const _Float16* __restrict__ WH,   // f16 weights in ws (or nullptr)
    float* __restrict__ out) {
    extern __shared__ char smem_raw[];
    _Float16* buf0 = (_Float16*)smem_raw;            // [13*16][PITCH]
    _Float16* buf1 = buf0 + BUF_HALVES;              // [13*16][PITCH]
    _Float16* hd   = buf1 + BUF_HALVES;              // [ 3*16][PITCH]
    float* scr  = (float*)(hd + HD_HALVES);          // [16][17] head dot products
    float* qbuf = scr + 16 * 17;                     // [48] x,y,t for 16 samples

    const int tid  = threadIdx.x;
    const int lane = tid & 31;
    const int wave = tid >> 5;
    const int n0   = blockIdx.x * SPW;

    // ---- Stage inputs ----
    if (tid < SPW) {
        qbuf[tid]      = x[n0 + tid];
        qbuf[16 + tid] = y[n0 + tid];
        qbuf[32 + tid] = t[n0 + tid];
    }
    __syncthreads();

    // ---- Layer 1 (3 -> 256), elementwise: input jet is the identity seed ----
    {
        const float w0 = W1[tid * 3 + 0], w1 = W1[tid * 3 + 1], w2 = W1[tid * 3 + 2];
        const float bb = b1[tid];
#pragma unroll 4
        for (int s = 0; s < SPW; ++s) {
            const float z  = w0 * qbuf[s] + w1 * qbuf[16 + s] + w2 * qbuf[32 + s] + bb;
            const float h  = fast_tanh(z);
            const float d1 = 1.f - h * h;
            const float d2 = -2.f * h * d1;
            const float d3 = d1 * (6.f * h * h - 2.f);
            float vv[13];
            vv[0]  = h;
            vv[1]  = d1 * w0;  vv[2] = d1 * w1;  vv[3] = d1 * w2;
            vv[4]  = d2 * w0 * w0;  vv[5] = d2 * w0 * w1;  vv[6] = d2 * w1 * w1;
            vv[7]  = d2 * w0 * w2;  vv[8] = d2 * w1 * w2;
            vv[9]  = d3 * w0 * w0 * w0;  vv[10] = d3 * w0 * w0 * w1;
            vv[11] = d3 * w0 * w1 * w1;  vv[12] = d3 * w1 * w1 * w1;
#pragma unroll
            for (int c = 0; c < 13; ++c)
                buf0[(c * 16 + s) * PITCH + tid] = (_Float16)vv[c];
        }
    }
    __syncthreads();

    // ---- Trunk layers 2,3 and both heads' hidden layers (WMMA GEMMs) ----
    const _Float16* Wh2  = F16W ? WH            : nullptr;
    const _Float16* Wh3  = F16W ? WH + WMAT     : nullptr;
    const _Float16* Whp1 = F16W ? WH + 2 * WMAT : nullptr;
    const _Float16* Whd1 = F16W ? WH + 3 * WMAT : nullptr;

    gemm_tanh_jet<13, F16W>(buf0, buf1, W2, Wh2, b2, lane, wave);
    __syncthreads();
    gemm_tanh_jet<13, F16W>(buf1, buf0, W3, Wh3, b3, lane, wave);   // trunk jet -> buf0
    __syncthreads();
    gemm_tanh_jet<13, F16W>(buf0, buf1, Wp1, Whp1, bp1, lane, wave); // pde head hidden -> buf1
    gemm_tanh_jet<3,  F16W>(buf0, hd,   Wd1, Whd1, bd1, lane, wave); // data head hidden -> hd
    __syncthreads();

    // ---- Final 2-wide output layers: 17 dot products per sample ----
    // j 0..11 : s-jet comps dx..dyyy       (Wp2 row 0)
    // j 12,13 : p_x, p_y                   (Wp2 row 1)
    // j 14    : p_pred value               (Wd2 row 1)
    // j 15,16 : sd_x, sd_y                 (Wd2 row 0)
    for (int job = tid; job < 17 * 16; job += 256) {
        const int s = job & 15;
        const int j = job >> 4;
        const _Float16* src;
        const float* wrow;
        int comp;
        if (j < 12)       { src = buf1; comp = j + 1;  wrow = Wp2; }
        else if (j < 14)  { src = buf1; comp = j - 11; wrow = Wp2 + HID; }
        else if (j == 14) { src = hd;   comp = 0;      wrow = Wd2 + HID; }
        else              { src = hd;   comp = j - 14; wrow = Wd2; }
        const v8h* row8 = (const v8h*)(src + (comp * 16 + s) * PITCH);
        float acc = 0.f;
#pragma unroll 4
        for (int o8 = 0; o8 < HID / 8; ++o8) {
            const v8h hv = row8[o8];                       // ds_load_b128
            const float4* w4 = (const float4*)(wrow + o8 * 8);
            const float4 wa = w4[0], wb = w4[1];
            acc += wa.x * (float)hv[0] + wa.y * (float)hv[1]
                 + wa.z * (float)hv[2] + wa.w * (float)hv[3]
                 + wb.x * (float)hv[4] + wb.y * (float)hv[5]
                 + wb.z * (float)hv[6] + wb.w * (float)hv[7];
        }
        scr[s * 17 + j] = acc;
    }
    __syncthreads();

    // ---- Assemble Navier-Stokes residuals + data-head predictions ----
    if (tid < SPW) {
        const float* S = scr + tid * 17;
        const float lam1 = *lam1p, lam2 = *lam2p;
        const float s_x = S[0],  s_y = S[1];
        const float s_xx = S[3], s_xy = S[4], s_yy = S[5], s_xt = S[6], s_yt = S[7];
        const float s_xxx = S[8], s_xxy = S[9], s_xyy = S[10], s_yyy = S[11];
        const float p_x = S[12], p_y = S[13];
        const float p_pred = S[14] + bd2[1];
        const float sd_x = S[15], sd_y = S[16];

        const float u_ = s_y, v_ = -s_x;
        const float u_x = s_xy,  u_y = s_yy,  u_t = s_yt;
        const float v_x = -s_xx, v_y = -s_xy, v_t = -s_xt;
        const float u_xx = s_xxy, u_yy = s_yyy;
        const float v_xx = -s_xxx, v_yy = -s_xyy;

        const float f_ = lam1 * (u_t + u_ * u_x + v_ * u_y) + p_x - lam2 * (u_xx + u_yy);
        const float g_ = lam1 * (v_t + u_ * v_x + v_ * v_y) + p_y - lam2 * (v_xx + v_yy);

        const int n = n0 + tid;
        out[0 * NPTS + n] = p_pred;   // p_pred
        out[1 * NPTS + n] = sd_y;     // u_pred =  d s_data / dy
        out[2 * NPTS + n] = -sd_x;    // v_pred = -d s_data / dx
        out[3 * NPTS + n] = f_;
        out[4 * NPTS + n] = g_;
    }
}

extern "C" void kernel_launch(void* const* d_in, const int* in_sizes, int n_in,
                              void* d_out, int out_size, void* d_ws, size_t ws_size,
                              hipStream_t stream) {
    (void)in_sizes; (void)n_in; (void)out_size;
    const float* x   = (const float*)d_in[0];
    const float* y   = (const float*)d_in[1];
    const float* t   = (const float*)d_in[2];
    // d_in[3..5] = p,u,v : unused by the reference outputs
    const float* W1  = (const float*)d_in[6];
    const float* b1  = (const float*)d_in[7];
    const float* W2  = (const float*)d_in[8];
    const float* b2  = (const float*)d_in[9];
    const float* W3  = (const float*)d_in[10];
    const float* b3  = (const float*)d_in[11];
    const float* Wp1 = (const float*)d_in[12];
    const float* bp1 = (const float*)d_in[13];
    const float* Wp2 = (const float*)d_in[14];
    // d_in[15] = bp2 : constants vanish in all required derivatives / unused values
    const float* Wd1 = (const float*)d_in[16];
    const float* bd1 = (const float*)d_in[17];
    const float* Wd2 = (const float*)d_in[18];
    const float* bd2 = (const float*)d_in[19];
    const float* l1  = (const float*)d_in[20];
    const float* l2  = (const float*)d_in[21];
    float* out = (float*)d_out;

    const bool use_f16w = (d_ws != nullptr) && (ws_size >= WS_NEEDED);

    if (use_f16w) {
        // Dynamic LDS > 64 KB needs the opt-in attribute (idempotent).
        hipFuncSetAttribute(reinterpret_cast<const void*>(pinn_jet_fused<true>),
                            hipFuncAttributeMaxDynamicSharedMemorySize, SMEM_BYTES);
        _Float16* wh = (_Float16*)d_ws;
        convert_weights<<<4 * WMAT / 256, 256, 0, stream>>>(W2, W3, Wp1, Wd1, wh);
        pinn_jet_fused<true><<<NPTS / SPW, 256, SMEM_BYTES, stream>>>(
            x, y, t, W1, b1, W2, b2, W3, b3, Wp1, bp1, Wp2, Wd1, bd1, Wd2, bd2,
            l1, l2, wh, out);
    } else {
        hipFuncSetAttribute(reinterpret_cast<const void*>(pinn_jet_fused<false>),
                            hipFuncAttributeMaxDynamicSharedMemorySize, SMEM_BYTES);
        pinn_jet_fused<false><<<NPTS / SPW, 256, SMEM_BYTES, stream>>>(
            x, y, t, W1, b1, W2, b2, W3, b3, Wp1, bp1, Wp2, Wd1, bd1, Wd2, bd2,
            l1, l2, nullptr, out);
    }
}
